// VGAE_8323646620419
// MI455X (gfx1250) — matrix operand order repaired and checked by reading, not compiled, not run on hardware
//
#include <hip/hip_runtime.h>
#include <hip/hip_bf16.h>

#define N_NODES 100000
#define N_EDGES 1600000
#define NFEAT   256
#define NHID    128
#define NCOLS   256   // fused [W_mean | W_logstd] columns

typedef __attribute__((ext_vector_type(16))) __bf16          v16bf;
typedef __attribute__((ext_vector_type(16))) unsigned short  v16us;
typedef __attribute__((ext_vector_type(8)))  float           v8f;

__device__ __forceinline__ unsigned short f32_to_bf16_rne(float f) {
    unsigned int u = __builtin_bit_cast(unsigned int, f);
    u += 0x7FFFu + ((u >> 16) & 1u);   // round-to-nearest-even
    return (unsigned short)(u >> 16);
}

// ---------------------------------------------------------------------------
// Kernel 1: pack [W_mean | W_logstd] -> bf16 fragments in WMMA B-layout.
// Fragment (c64, k32q, nc): lane l holds K = c64*64 + k32q*32 + 16*(l>=16) + i,
// col = nc*16 + (l&15), i = 0..15, stored as contiguous 32 B per lane.
// ---------------------------------------------------------------------------
__global__ __launch_bounds__(256) void k_prep_w(const float* __restrict__ Wm,
                                                const float* __restrict__ Wl,
                                                unsigned short* __restrict__ WcFrag) {
    int tid  = blockIdx.x * 256 + threadIdx.x;  // 4096 total
    int lane = tid & 31;
    int g    = tid >> 5;                        // fragment id 0..127
    int c64  = g >> 5;                          // 64-row K chunk
    int fic  = g & 31;                          // fragment in chunk = k32q*16 + nc
    int k32q = fic >> 4;
    int nc   = fic & 15;
    int kbase = (lane >> 4) * 16;
    int col   = nc * 16 + (lane & 15);

    v16us u;
#pragma unroll
    for (int i = 0; i < 16; ++i) {
        int k = c64 * 64 + k32q * 32 + kbase + i;
        float v = (col < NHID) ? Wm[k * NHID + col] : Wl[k * NHID + (col - NHID)];
        u[i] = f32_to_bf16_rne(v);
    }
    ((v16us*)WcFrag)[tid] = u;
}

// ---------------------------------------------------------------------------
// Kernel 2: convert x (f32) -> bf16 fragments in WMMA A-layout.
// Fragment (rs, kq): lane l holds M = rs*16 + (l&15),
// K = kq*32 + 16*(l>=16) + i, 32 B contiguous per lane.
// ---------------------------------------------------------------------------
__global__ __launch_bounds__(256) void k_prep_x(const float* __restrict__ x,
                                                unsigned short* __restrict__ xA) {
    int tid  = blockIdx.x * 256 + threadIdx.x;  // (N/16)*8*32 = 1,600,000 exact
    int lane = tid & 31;
    int g    = tid >> 5;                        // rs*8 + kq
    int kq   = g & 7;
    int rs   = g >> 3;
    int m    = lane & 15;
    int kbaseA = (lane >> 4) * 16;

    const float* xr = x + (size_t)(rs * 16 + m) * NFEAT + kq * 32 + kbaseA;
    v16us u;
#pragma unroll
    for (int i = 0; i < 16; ++i)
        u[i] = f32_to_bf16_rne(xr[i]);
    ((v16us*)xA)[tid] = u;
}

// ---------------------------------------------------------------------------
// Kernel 3: zero accumulator + KL output slot
// ---------------------------------------------------------------------------
__global__ __launch_bounds__(256) void k_zero(float4* __restrict__ accum4,
                                              size_t count4,
                                              float* __restrict__ kld_slot) {
    size_t i = (size_t)blockIdx.x * blockDim.x + threadIdx.x;
    size_t stride = (size_t)gridDim.x * blockDim.x;
    float4 z = make_float4(0.f, 0.f, 0.f, 0.f);
    for (; i < count4; i += stride) accum4[i] = z;
    if (blockIdx.x == 0 && threadIdx.x == 0) *kld_slot = 0.f;
}

// ---------------------------------------------------------------------------
// Kernel 4: support[N,256] = x @ Wc via v_wmma_f32_16x16x32_bf16.
// 8 waves = 2 row-strips x 4 col-groups; block tile 32x256.
// W staged per 64-K chunk (32 KB) in LDS as pre-swizzled fragments:
// B-frag read = 2x ds_load_b128, A-frag read = 2x global_load_b128.
// ---------------------------------------------------------------------------
__global__ __launch_bounds__(256) void k_gemm(const unsigned short* __restrict__ xA,
                                              const unsigned short* __restrict__ WcFrag,
                                              float* __restrict__ support) {
    __shared__ __align__(16) unsigned short ldsW[16384];   // 32 KB = one 64-K chunk

    const int tid    = threadIdx.x;
    const int lane   = tid & 31;
    const int wave   = tid >> 5;
    const int rstrip = wave & 1;                // 0..1
    const int cgroup = wave >> 1;               // 0..3
    const int rs     = blockIdx.x * 2 + rstrip; // 16-row strip index
    const int m      = lane & 15;
    const int hi     = lane >> 4;

    v8f acc[4] = {v8f{}, v8f{}, v8f{}, v8f{}};

    const v16us* Af = (const v16us*)xA;
    const v16us* Bf = (const v16us*)ldsW;

    for (int c64 = 0; c64 < 4; ++c64) {
        // Cooperative contiguous copy of one 32 KB fragment chunk into LDS.
        const uint4* gsrc = (const uint4*)(WcFrag + c64 * 16384);
        uint4*       ldst = (uint4*)ldsW;
#pragma unroll
        for (int i = 0; i < 8; ++i) ldst[tid + i * 256] = gsrc[tid + i * 256];
        __syncthreads();

#pragma unroll
        for (int kq2 = 0; kq2 < 2; ++kq2) {
            v16bf a = __builtin_bit_cast(
                v16bf, Af[((size_t)rs * 8 + c64 * 2 + kq2) * 32 + lane]);
#pragma unroll
            for (int t = 0; t < 4; ++t) {
                int nc = cgroup * 4 + t;
                v16bf b = __builtin_bit_cast(v16bf, Bf[(kq2 * 16 + nc) * 32 + lane]);
                acc[t] = __builtin_amdgcn_wmma_f32_16x16x32_bf16(
                    /*neg_a=*/false, a, /*neg_b=*/false, b,
                    /*c_mod=*/(short)0, acc[t],
                    /*reuse_a=*/false, /*reuse_b=*/false);
            }
        }
        __syncthreads();
    }

    // C/D layout: VGPR r, lane -> row = r + 8*hi, col = lane&15
#pragma unroll
    for (int t = 0; t < 4; ++t) {
#pragma unroll
        for (int r = 0; r < 8; ++r) {
            int row = rs * 16 + r + hi * 8;
            int col = (cgroup * 4 + t) * 16 + m;
            support[(size_t)row * NCOLS + col] = acc[t][r];
        }
    }
}

// ---------------------------------------------------------------------------
// Kernel 5: SpMM scatter-add. One wave per edge; lane covers 8 fused features
// via two float4 gathers. support (102 MB) fits in 192 MB L2.
// ---------------------------------------------------------------------------
__global__ __launch_bounds__(256) void k_spmm(const int*   __restrict__ erow,
                                              const int*   __restrict__ ecol,
                                              const float* __restrict__ eval,
                                              const float* __restrict__ support,
                                              float*       __restrict__ accum) {
    int e = blockIdx.x * 8 + (threadIdx.x >> 5);
    if (e >= N_EDGES) return;
    int lane = threadIdx.x & 31;
    int row = erow[e];
    int col = ecol[e];
    float v = eval[e];

    const float4* src = (const float4*)(support + (size_t)col * NCOLS);
    float4 s0 = src[lane];
    float4 s1 = src[lane + 32];

    float* dst = accum + (size_t)row * NCOLS;
    int f0 = 4 * lane;
    atomicAdd(dst + f0 + 0,   v * s0.x);
    atomicAdd(dst + f0 + 1,   v * s0.y);
    atomicAdd(dst + f0 + 2,   v * s0.z);
    atomicAdd(dst + f0 + 3,   v * s0.w);
    atomicAdd(dst + f0 + 128, v * s1.x);
    atomicAdd(dst + f0 + 129, v * s1.y);
    atomicAdd(dst + f0 + 130, v * s1.z);
    atomicAdd(dst + f0 + 131, v * s1.w);
}

// ---------------------------------------------------------------------------
// Kernel 6: bias + reparameterize + KL reduction.
// kld = sum_{n,j} kl / N   (mean over nodes, sum over features)
// ---------------------------------------------------------------------------
__global__ __launch_bounds__(256) void k_final(const float* __restrict__ accum,
                                               const float* __restrict__ b_mean,
                                               const float* __restrict__ b_logstd,
                                               const float* __restrict__ noise,
                                               float* __restrict__ z_out,
                                               float* __restrict__ kld_out) {
    __shared__ float red[256];
    const int tid = threadIdx.x;
    size_t idx = (size_t)blockIdx.x * 256 + tid;    // over N*128 exactly

    int n = (int)(idx >> 7);
    int j = (int)(idx & 127);
    float mean   = accum[(size_t)n * NCOLS + j]        + b_mean[j];
    float logstd = accum[(size_t)n * NCOLS + NHID + j] + b_logstd[j];

    float ex = expf(logstd);
    z_out[idx] = noise[idx] * ex + mean;

    float sp = (ex > 20.f) ? ex : log1pf(expf(ex));          // softplus, safe
    float sigma1 = 0.1f + 0.9f * sp;
    const float sigma2 = 0.1f + 0.9f * log1pf(expf(2.718281828f)); // softplus(e)
    float kl = logf(sigma2 / sigma1)
             + (sigma1 * sigma1 + mean * mean) / (2.f * sigma2 * sigma2)
             - 0.5f;

    red[tid] = kl;
    __syncthreads();
#pragma unroll
    for (int s = 128; s > 0; s >>= 1) {
        if (tid < s) red[tid] += red[tid + s];
        __syncthreads();
    }
    if (tid == 0) atomicAdd(kld_out, red[0] * (1.0f / (float)N_NODES));
}

// ---------------------------------------------------------------------------
extern "C" void kernel_launch(void* const* d_in, const int* in_sizes, int n_in,
                              void* d_out, int out_size, void* d_ws, size_t ws_size,
                              hipStream_t stream) {
    const float* x     = (const float*)d_in[0];
    const int*   erow  = (const int*)  d_in[1];
    const int*   ecol  = (const int*)  d_in[2];
    const float* eval  = (const float*)d_in[3];
    const float* Wm    = (const float*)d_in[4];
    const float* bm    = (const float*)d_in[5];
    const float* Wl    = (const float*)d_in[6];
    const float* bl    = (const float*)d_in[7];
    const float* noise = (const float*)d_in[8];

    float* out = (float*)d_out;
    float* kld = out + (size_t)N_NODES * NHID;      // last element

    // ws: WcFrag 128 KB | accum 102.4 MB | support 102.4 MB | xA 51.2 MB
    char* ws = (char*)d_ws;
    unsigned short* WcFrag = (unsigned short*)ws;
    float* accum   = (float*)(ws + 131072);
    float* support = (float*)(ws + 131072 + (size_t)N_NODES * NCOLS * sizeof(float));
    unsigned short* xA =
        (unsigned short*)(ws + 131072 + 2 * (size_t)N_NODES * NCOLS * sizeof(float));

    k_prep_w<<<16, 256, 0, stream>>>(Wm, Wl, WcFrag);
    k_prep_x<<<(N_NODES / 16) * 8 * 32 / 256, 256, 0, stream>>>(x, xA);
    k_zero<<<2048, 256, 0, stream>>>((float4*)accum, (size_t)N_NODES * NCOLS / 4, kld);
    k_gemm<<<N_NODES / 32, 256, 0, stream>>>(xA, WcFrag, support);
    k_spmm<<<(N_EDGES + 7) / 8, 256, 0, stream>>>(erow, ecol, eval, support, accum);
    k_final<<<(N_NODES * NHID) / 256, 256, 0, stream>>>(accum, bm, bl, noise, out, kld);
}
